// RSTRM_70300024701416
// MI455X (gfx1250) — matrix operand order repaired and checked by reading, not compiled, not run on hardware
//
#include <hip/hip_runtime.h>
#include <hip/hip_bf16.h>

typedef __attribute__((ext_vector_type(16))) __bf16 v16bf;
typedef __attribute__((ext_vector_type(8)))  float  v8f;
typedef __attribute__((ext_vector_type(4)))  unsigned int v4u;
typedef __attribute__((ext_vector_type(8)))  int v8i;
typedef __attribute__((ext_vector_type(4)))  int v4i;

#define B_   128
#define D_   512
#define S_   512
#define KTOP 10

// ---------------- transpose: x[b,d,s] -> feat[b,s,d] ----------------
__global__ void k_transpose(const float* __restrict__ x, float* __restrict__ feat) {
  __shared__ float tile[32][33];
  int b = blockIdx.z;
  int s0 = blockIdx.x * 32, d0 = blockIdx.y * 32;
  int tx = threadIdx.x, ty = threadIdx.y; // 32x8
  const float* xb = x + (size_t)b * D_ * S_;
  float* fb = feat + (size_t)b * S_ * D_;
#pragma unroll
  for (int j = 0; j < 4; ++j)
    tile[ty + j * 8][tx] = xb[(size_t)(d0 + ty + j * 8) * S_ + (s0 + tx)];
  __syncthreads();
#pragma unroll
  for (int j = 0; j < 4; ++j)
    fb[(size_t)(s0 + ty + j * 8) * D_ + (d0 + tx)] = tile[tx][ty + j * 8];
}

// ---------------- info_g[b,d] = mean_s x[b,d,s] (wave per row) ----------------
__global__ void k_rowmean(const float* __restrict__ x, float* __restrict__ infog) {
  int row = blockIdx.x * 8 + (threadIdx.x >> 5);
  int lane = threadIdx.x & 31;
  const float* p = x + (size_t)row * S_;
  float s = 0.f;
  for (int i = lane; i < S_; i += 32) s += p[i];
#pragma unroll
  for (int off = 16; off; off >>= 1) s += __shfl_xor(s, off, 32);
  if (lane == 0) infog[row] = s * (1.0f / S_);
}

// ---- split+transpose top half of w1: w1t_{hi,lo}[n][k] = bf16split(w1[k][n]) ----
__global__ void k_split_w1t(const float* __restrict__ w1, __bf16* __restrict__ hi,
                            __bf16* __restrict__ lo) {
  __shared__ float tile[32][33];
  int k0 = blockIdx.x * 32, n0 = blockIdx.y * 32;
  int tx = threadIdx.x, ty = threadIdx.y; // 32x8
#pragma unroll
  for (int j = 0; j < 4; ++j)
    tile[ty + j * 8][tx] = w1[(size_t)(k0 + ty + j * 8) * D_ + (n0 + tx)];
  __syncthreads();
#pragma unroll
  for (int j = 0; j < 4; ++j) {
    float v = tile[tx][ty + j * 8];
    size_t o = (size_t)(n0 + ty + j * 8) * D_ + (k0 + tx);
    __bf16 h = (__bf16)v;
    hi[o] = h;
    lo[o] = (__bf16)(v - (float)h);
  }
}

// ---------------- pew1[s,d] = b1[d] + sum_f PE[s,f]*w1[f,d] ----------------
__global__ void k_pew1(const float* __restrict__ w1, const float* __restrict__ b1,
                       float* __restrict__ pew1) {
  __shared__ float pe[2 * D_];
  int s = blockIdx.x, tid = threadIdx.x;
  for (int i = tid; i < D_; i += 256) {
    float freq = expf((float)i * (-9.210340371976184f / 512.0f)); // exp(-(2i)ln1e4/1024)
    float ang = (float)s * freq;
    pe[2 * i] = sinf(ang);
    pe[2 * i + 1] = cosf(ang);
  }
  __syncthreads();
  for (int d = tid; d < D_; d += 256) {
    float acc = b1[d];
    for (int f = 0; f < 2 * D_; ++f) acc = fmaf(pe[f], w1[(size_t)f * D_ + d], acc);
    pew1[(size_t)s * D_ + d] = acc;
  }
}

// ---------------- meanw[b,d] = sum_f infog[b,f]*w1[512+f,d] ----------------
__global__ void k_meanw(const float* __restrict__ infog, const float* __restrict__ w1,
                        float* __restrict__ meanw) {
  __shared__ float ig[D_];
  int b = blockIdx.x, tid = threadIdx.x;
  for (int i = tid; i < D_; i += 256) ig[i] = infog[(size_t)b * D_ + i];
  __syncthreads();
  const float* wbot = w1 + (size_t)D_ * D_;
  for (int d = tid; d < D_; d += 256) {
    float acc = 0.f;
    for (int f = 0; f < D_; ++f) acc = fmaf(ig[f], wbot[(size_t)f * D_ + d], acc);
    meanw[(size_t)b * D_ + d] = acc;
  }
}

// ---------------- WMMA operand builders (ISA 7.12.2 layouts, wave32) ----------------
__device__ __forceinline__ v16bf ld_a(const __bf16* row, int abase) {
  v16bf a;
#pragma unroll
  for (int e = 0; e < 8; ++e) a[e] = row[abase + e];
#pragma unroll
  for (int e = 0; e < 8; ++e) a[8 + e] = row[16 + abase + e];
  return a;
}
__device__ __forceinline__ v16bf ld_b(const __bf16* row, int kbase) {
  v16bf bv;
#pragma unroll
  for (int e = 0; e < 8; ++e) bv[e] = row[kbase + e];
#pragma unroll
  for (int e = 0; e < 8; ++e) bv[8 + e] = row[16 + kbase + e];
  return bv;
}

// TDM: pull a 64-row x 64-byte tile of w1t (row stride 1024B) into LDS,
// padded by hardware to an 80B LDS row stride ([64][40] bf16).
// D# fields per CDNA5 ISA (2D tensor, groups 2/3 zero).
__device__ __forceinline__ void tdm_load_tile(unsigned lds_off, const __bf16* gbase,
                                              int n0, int k0) {
  unsigned long long ga =
      (unsigned long long)(size_t)gbase + (unsigned)(n0 * (D_ * 2) + k0 * 2);
  v4u g0;
  g0[0] = 1u;                                  // count=1, user descriptor
  g0[1] = lds_off;                             // lds_addr (bytes)
  g0[2] = (unsigned)ga;                        // global_addr[31:0]
  g0[3] = (unsigned)(ga >> 32) | (2u << 30);   // global_addr[56:32] | type=2
  v8i g1;
  g1[0] = (3 << 16)      // data_size = 8B units
        | (1 << 20)      // pad_enable
        | (3 << 22)      // pad_interval: every 16 DWORDs (64B)
        | (3 << 25);     // pad_amount: 4 DWORDs (16B)
  g1[1] = (128 << 16);   // tensor_dim0 = 128 (x8B = 1024B row)
  g1[2] = (512 << 16);   // tensor_dim1 = 512 rows
  g1[3] = (8 << 16);     // tile_dim0 = 8 (x8B = 64B)
  g1[4] = 64;            // tile_dim1 = 64 rows
  g1[5] = 128;           // tensor_dim0_stride = 128 (x8B)
  g1[6] = 0;
  g1[7] = 0;
  v4i z4 = {0, 0, 0, 0};
  v8i z8 = {0, 0, 0, 0, 0, 0, 0, 0};
  __builtin_amdgcn_tensor_load_to_lds(g0, g1, z4, z4, z8, 0);
}

// ---------------- fused GEMM + epilogue -> score[b,s] ----------------
// WG = 256 thr (8 waves). A stripe (64 x 512, bf16 hi/lo) resident in LDS.
// B tiles (64 x 32 bf16 hi/lo) double-buffered via Tensor Data Mover.
// acc = A_hi*B_hi + A_hi*B_lo + A_lo*B_hi  (bf16-split ~fp32 accuracy)
__global__ __launch_bounds__(256) void k_gemm_score(
    const float* __restrict__ feat, const __bf16* __restrict__ w1h,
    const __bf16* __restrict__ w1l, const float* __restrict__ pew1,
    const float* __restrict__ meanw, const float* __restrict__ w2,
    const float* __restrict__ b2, float* __restrict__ score) {
  extern __shared__ char smem[];
  // LDS layout (bytes):
  //   AsH : 64*520*2 = 66560   @ 0
  //   AsL : 66560              @ 66560
  //   Bs  : 4 sub-buffers (buf0h, buf0l, buf1h, buf1l) x 64*40*2 = 5120 @ 133120
  //   sred: 64 f32             @ 153600
  __bf16* AsH = (__bf16*)smem;
  __bf16* AsL = (__bf16*)(smem + 66560);
  __bf16* Bs  = (__bf16*)(smem + 133120);
  float* sred = (float*)(smem + 153600);
  unsigned bs_lds = (unsigned)(size_t)(void*)Bs; // LDS byte offset (addr[31:0])

  int b = blockIdx.y;
  int s0 = blockIdx.x * 64;
  int tid = threadIdx.x;
  int lane = tid & 31, wave = tid >> 5;
  int wm = wave & 3, wn = wave >> 2; // 4 waves in M, 2 in N
  const float* fb = feat + ((size_t)b * S_ + s0) * D_;

  // prologue: wave 0 kicks off TDM for tile t=0 (overlaps with A staging)
  if (tid < 32) {
    tdm_load_tile(bs_lds + 0 * 5120, w1h, 0, 0);
    tdm_load_tile(bs_lds + 1 * 5120, w1l, 0, 0);
  }
  if (tid < 64) sred[tid] = 0.f;

  // stage full A stripe once: 64 x 512 f32 -> bf16 hi/lo, row stride 520
  for (int i = tid; i < 64 * D_; i += 256) {
    int r = i >> 9, c = i & (D_ - 1);
    float v = fb[(size_t)r * D_ + c];
    __bf16 h = (__bf16)v;
    AsH[r * 520 + c] = h;
    AsL[r * 520 + c] = (__bf16)(v - (float)h);
  }

  float sp[8];
#pragma unroll
  for (int r = 0; r < 8; ++r) sp[r] = 0.f;

  int am = 16 * wm + (lane & 15);
  int abase = (lane >> 4) * 8;
  int kbase = (lane >> 4) * 8;
  int nl0 = 32 * wn + (lane & 15);
  int nl1 = nl0 + 16;

  for (int nc = 0; nc < 8; ++nc) {
    int n0 = nc * 64;
    v8f acc0 = {}; v8f acc1 = {};
    for (int kc = 0; kc < 16; ++kc) {
      int t = nc * 16 + kc;
      int buf = t & 1;
      __syncthreads(); // previous compute on buf^1 done -> safe to DMA into it
      if (tid < 32) {
        if (t + 1 < 128) {
          int t1 = t + 1, nc1 = t1 >> 4, kc1 = t1 & 15;
          tdm_load_tile(bs_lds + (unsigned)(((buf ^ 1) * 2 + 0) * 5120), w1h,
                        nc1 * 64, kc1 * 32);
          tdm_load_tile(bs_lds + (unsigned)(((buf ^ 1) * 2 + 1) * 5120), w1l,
                        nc1 * 64, kc1 * 32);
          __builtin_amdgcn_s_wait_tensorcnt(2); // tile t landed (2 newest pending)
        } else {
          __builtin_amdgcn_s_wait_tensorcnt(0);
        }
      }
      __syncthreads(); // tile t visible to all waves

      int k0 = kc * 32;
      const __bf16* arH = AsH + am * 520 + k0;
      const __bf16* arL = AsL + am * 520 + k0;
      const __bf16* bH = Bs + (buf * 2 + 0) * 2560;
      const __bf16* bL = Bs + (buf * 2 + 1) * 2560;
      v16bf ah = ld_a(arH, abase);
      v16bf al = ld_a(arL, abase);
      v16bf bh0 = ld_b(bH + nl0 * 40, kbase);
      v16bf bl0 = ld_b(bL + nl0 * 40, kbase);
      v16bf bh1 = ld_b(bH + nl1 * 40, kbase);
      v16bf bl1 = ld_b(bL + nl1 * 40, kbase);
      acc0 = __builtin_amdgcn_wmma_f32_16x16x32_bf16(false, ah, false, bh0, (short)0, acc0, false, false);
      acc1 = __builtin_amdgcn_wmma_f32_16x16x32_bf16(false, ah, false, bh1, (short)0, acc1, false, false);
      acc0 = __builtin_amdgcn_wmma_f32_16x16x32_bf16(false, ah, false, bl0, (short)0, acc0, false, false);
      acc1 = __builtin_amdgcn_wmma_f32_16x16x32_bf16(false, ah, false, bl1, (short)0, acc1, false, false);
      acc0 = __builtin_amdgcn_wmma_f32_16x16x32_bf16(false, al, false, bh0, (short)0, acc0, false, false);
      acc1 = __builtin_amdgcn_wmma_f32_16x16x32_bf16(false, al, false, bh1, (short)0, acc1, false, false);
    }
    // epilogue: + pew1 + meanw, ReLU, dot with w2 -> per-row partials
    int col0 = n0 + nl0, col1 = n0 + nl1;
    int rbase = 16 * wm + (lane >> 4) * 8;
    float m0 = meanw[(size_t)b * D_ + col0];
    float m1 = meanw[(size_t)b * D_ + col1];
    float w20 = w2[col0], w21 = w2[col1];
#pragma unroll
    for (int r = 0; r < 8; ++r) {
      int srow = s0 + rbase + r;
      float h0 = acc0[r] + pew1[(size_t)srow * D_ + col0] + m0;
      float h1 = acc1[r] + pew1[(size_t)srow * D_ + col1] + m1;
      h0 = h0 > 0.f ? h0 : 0.f;
      h1 = h1 > 0.f ? h1 : 0.f;
      sp[r] += h0 * w20 + h1 * w21;
    }
  }
  // reduce over the 16 lanes of each half-wave (xor 1..8 stays inside half)
#pragma unroll
  for (int r = 0; r < 8; ++r) {
    float v = sp[r];
#pragma unroll
    for (int off = 8; off; off >>= 1) v += __shfl_xor(v, off, 32);
    sp[r] = v;
  }
  if ((lane & 15) == 0) {
    int rbase = 16 * wm + (lane >> 4) * 8;
#pragma unroll
    for (int r = 0; r < 8; ++r) atomicAdd(&sred[rbase + r], sp[r]);
  }
  __syncthreads();
  if (tid < 64) score[(size_t)b * S_ + s0 + tid] = sred[tid] + b2[0];
}

// ---------------- per-batch top-10 -> one-hot indices + gathered features ----------------
__global__ void k_topk(const float* __restrict__ score, const float* __restrict__ feat,
                       float* __restrict__ ind, float* __restrict__ sel) {
  __shared__ float vals[S_];
  __shared__ float rv[256];
  __shared__ int ri[256];
  __shared__ int top[KTOP];
  int b = blockIdx.x, tid = threadIdx.x;
  vals[tid] = score[(size_t)b * S_ + tid];
  vals[tid + 256] = score[(size_t)b * S_ + tid + 256];
  __syncthreads();
  for (int k = 0; k < KTOP; ++k) {
    float v0 = vals[tid], v1 = vals[tid + 256];
    int i0 = tid;
    if (v1 > v0) { v0 = v1; i0 = tid + 256; } // tie -> lower index (jax top_k)
    rv[tid] = v0; ri[tid] = i0;
    __syncthreads();
    for (int sr = 128; sr > 0; sr >>= 1) {
      if (tid < sr) {
        float a = rv[tid], c = rv[tid + sr];
        int ia = ri[tid], ic = ri[tid + sr];
        if (c > a || (c == a && ic < ia)) { rv[tid] = c; ri[tid] = ic; }
      }
      __syncthreads();
    }
    if (tid == 0) { top[k] = ri[0]; vals[ri[0]] = -__builtin_inff(); }
    __syncthreads();
  }
  if (tid == 0) { // sort indices ascending
    for (int i = 1; i < KTOP; ++i) {
      int key = top[i], j = i - 1;
      while (j >= 0 && top[j] > key) { top[j + 1] = top[j]; --j; }
      top[j + 1] = key;
    }
  }
  __syncthreads();
  for (int i = tid; i < KTOP * S_; i += 256) {
    int k = i >> 9, s = i & (S_ - 1);
    ind[(size_t)b * KTOP * S_ + i] = (s == top[k]) ? 1.0f : 0.0f;
  }
  for (int i = tid; i < KTOP * D_; i += 256) {
    int k = i >> 9, d = i & (D_ - 1);
    sel[(size_t)b * KTOP * D_ + i] = feat[((size_t)b * S_ + top[k]) * D_ + d];
  }
}

extern "C" void kernel_launch(void* const* d_in, const int* in_sizes, int n_in,
                              void* d_out, int out_size, void* d_ws, size_t ws_size,
                              hipStream_t stream) {
  (void)in_sizes; (void)n_in; (void)out_size; (void)ws_size;
  const float* x  = (const float*)d_in[0];   // [128,512,512]
  const float* w1 = (const float*)d_in[1];   // [1024,512]
  const float* b1 = (const float*)d_in[2];   // [512]
  const float* w2 = (const float*)d_in[3];   // [512,1]
  const float* b2 = (const float*)d_in[4];   // [1]

  float* ind  = (float*)d_out;                       // [B,10,S]
  float* sel  = ind + (size_t)B_ * KTOP * S_;        // [B,10,D]
  float* feat = sel + (size_t)B_ * KTOP * D_;        // [B,S,D]

  float* score = (float*)d_ws;                 // B*S
  float* meanw = score + (size_t)B_ * S_;      // B*D
  float* pew1  = meanw + (size_t)B_ * D_;      // S*D
  float* infog = pew1 + (size_t)S_ * D_;       // B*D
  __bf16* w1h  = (__bf16*)(infog + (size_t)B_ * D_); // D*D bf16, transposed [n][k]
  __bf16* w1l  = w1h + (size_t)D_ * D_;              // D*D bf16, transposed [n][k]

  k_transpose<<<dim3(S_ / 32, D_ / 32, B_), dim3(32, 8), 0, stream>>>(x, feat);
  k_rowmean<<<B_ * D_ / 8, 256, 0, stream>>>(x, infog);
  k_split_w1t<<<dim3(D_ / 32, D_ / 32), dim3(32, 8), 0, stream>>>(w1, w1h, w1l);
  k_pew1<<<S_, 256, 0, stream>>>(w1, b1, pew1);
  k_meanw<<<B_, 256, 0, stream>>>(infog, w1, meanw);
  size_t smem_bytes = 153600 + 256; // A(hi/lo) + B double buffers + sred
  k_gemm_score<<<dim3(S_ / 64, B_), 256, smem_bytes, stream>>>(
      feat, w1h, w1l, pew1, meanw, w2, b2, score);
  k_topk<<<B_, 256, 0, stream>>>(score, feat, ind, sel);
}
